// MultiHeadSelfAttention_50422916055217
// MI455X (gfx1250) — matrix operand order repaired and checked by reading, not compiled, not run on hardware
//
#include <hip/hip_runtime.h>
#include <hip/hip_bf16.h>
#include <math.h>

// MI455X (gfx1250) flash-attention MHA.
// All matmuls: v_wmma_f32_16x16x32_bf16 (f32 accum).
// K/V tiles staged into LDS by the Tensor Data Mover (tensor_load_to_lds),
// double-buffered, synchronized with s_wait_tensorcnt + workgroup barriers.

typedef __bf16 bf16_t;
typedef __attribute__((ext_vector_type(16))) __bf16 v16bf;
typedef __attribute__((ext_vector_type(8)))  __bf16 v8bf;
typedef __attribute__((ext_vector_type(8)))  float  v8f;
typedef __attribute__((ext_vector_type(4))) unsigned int u32x4;
typedef __attribute__((ext_vector_type(8)))  int i32x8;
typedef __attribute__((ext_vector_type(4)))  int i32x4;

#define HEADS  8
#define DHEAD  128
#define BATCH  4
#define SEQ    2048
#define ROWS   (BATCH*SEQ)     // 8192
#define DMODEL (HEADS*DHEAD)   // 1024

#if defined(__has_builtin)
#  if __has_builtin(__builtin_amdgcn_tensor_load_to_lds) && \
      __has_builtin(__builtin_amdgcn_s_wait_tensorcnt)
#    define USE_TDM 1
#  endif
#endif
#ifndef USE_TDM
#  define USE_TDM 0
#endif

// ---- WMMA operand loaders (contiguous k order, same for A and B, wave32) ----
// A (16x32 bf16): lanes 0-15 hold row M=lane, K = c..c+15;
//                 lanes 16-31 hold row M=lane-16, K = c+16..c+31.
static __device__ inline v16bf load_a16(const bf16_t* __restrict__ base, int ld) {
  int lane = threadIdx.x & 31;
  int half = lane >> 4;
  int r    = lane & 15;
  const bf16_t* p = base + r * ld + half * 16;
  v8bf lo = *(const v8bf*)(p);
  v8bf hi = *(const v8bf*)(p + 8);
  v16bf out;
#pragma unroll
  for (int i = 0; i < 8; ++i) { out[i] = lo[i]; out[8 + i] = hi[i]; }
  return out;
}
// B (32x16 bf16) from a row-major N x K buffer: lane holds column N=lane%16,
// K-range = half*16 .. +15 (contiguous 32B per lane).
static __device__ inline v16bf load_b16(const bf16_t* __restrict__ base, int ld) {
  int lane = threadIdx.x & 31;
  int half = lane >> 4;
  int c    = lane & 15;
  return *(const v16bf*)(base + c * ld + half * 16);
}
static __device__ inline v8f wmma_bf16(v16bf a, v16bf b, v8f c) {
  return __builtin_amdgcn_wmma_f32_16x16x32_bf16(false, a, false, b, (short)0, c,
                                                 false, false);
}

#if USE_TDM
// Issue one TDM 2-D tile load (elements are 2 bytes). Tiles are fully
// in-bounds, so tensor dims == tile dims. D# layout per CDNA5 ISA §8.
static __device__ inline void tdm_load_2d(unsigned lds_addr, const void* gaddr,
                                          unsigned tile_d0, unsigned tile_d1,
                                          unsigned long long stride0_elems) {
  unsigned long long ga = (unsigned long long)(uintptr_t)gaddr;
  u32x4 g0;
  g0[0] = 1u;                                           // count=1, user D#
  g0[1] = lds_addr;                                     // LDS byte address
  g0[2] = (unsigned)(ga & 0xffffffffu);                 // global_addr[31:0]
  g0[3] = (unsigned)((ga >> 32) & 0x01ffffffu) | (2u << 30);  // addr[56:32], type=2
  i32x8 g1;
  g1[0] = (int)(1u << 16);                              // data_size=1 (2B), mask=0
  g1[1] = (int)((tile_d0 & 0xffffu) << 16);             // tensor_dim0[15:0]
  g1[2] = (int)(((tile_d0 >> 16) & 0xffffu) | ((tile_d1 & 0xffffu) << 16));
  g1[3] = (int)(((tile_d1 >> 16) & 0xffffu) | ((tile_d0 & 0xffffu) << 16)); // tile_dim0
  g1[4] = (int)(tile_d1 & 0xffffu);                     // tile_dim1 (tile_dim2=0)
  g1[5] = (int)(stride0_elems & 0xffffffffu);           // dim0 stride [31:0]
  g1[6] = (int)((stride0_elems >> 32) & 0xffffu);       // dim0 stride [47:32]
  g1[7] = 0;
  i32x4 z4 = {0, 0, 0, 0};
#if defined(__clang_major__) && (__clang_major__ >= 23)
  i32x8 z8 = {0, 0, 0, 0, 0, 0, 0, 0};
  __builtin_amdgcn_tensor_load_to_lds(g0, g1, z4, z4, z8, 0);
#else
  __builtin_amdgcn_tensor_load_to_lds(g0, g1, z4, z4, 0);
#endif
}
static __device__ inline unsigned lds_off(const void* p) {
  return (unsigned)(uintptr_t)p;   // flat LDS addr truncates to LDS offset
}
#endif

// ---------------- conversion kernels ----------------
__global__ __launch_bounds__(256) void k_cvt_x(const float* __restrict__ x,
                                               bf16_t* __restrict__ xb, int n) {
  int i = blockIdx.x * 256 + threadIdx.x;
  if (i < n) xb[i] = (bf16_t)x[i];
}

// WT[proj][n][d] = W[d][n] * (proj<2 ? 128^-0.25 : 1)   (proj: 0=q,1=k,2=v)
__global__ __launch_bounds__(256) void k_cvt_w(const float* __restrict__ Wq,
                                               const float* __restrict__ Wk,
                                               const float* __restrict__ Wv,
                                               bf16_t* __restrict__ wt) {
  int i = blockIdx.x * 256 + threadIdx.x;       // 3*1024*128
  int proj = i >> 17;
  int rem  = i & 131071;
  int n = rem >> 7;
  int d = rem & 127;
  const float* W = (proj == 0) ? Wq : (proj == 1) ? Wk : Wv;
  float s = (proj < 2) ? 0.29730177875068026f : 1.0f;   // 128^-0.25
  wt[i] = (bf16_t)(W[d * DMODEL + n] * s);
}

// WoT[n][k] = Wo[k][n],  n in [0,128), k in [0,1024)
__global__ __launch_bounds__(256) void k_cvt_wo(const float* __restrict__ Wo,
                                                bf16_t* __restrict__ wot) {
  int i = blockIdx.x * 256 + threadIdx.x;
  int n = i >> 10;
  int kk = i & 1023;
  wot[i] = (bf16_t)(Wo[kk * DHEAD + n]);
}

// ---------------- QKV projection GEMM ----------------
__global__ __launch_bounds__(256) void k_qkv(const bf16_t* __restrict__ xb,
                                             const bf16_t* __restrict__ wt,
                                             bf16_t* __restrict__ qb,
                                             bf16_t* __restrict__ kb,
                                             bf16_t* __restrict__ vtb) {
  int wave = (blockIdx.x * 256 + threadIdx.x) >> 5;
  int proj = wave >> 13;
  int r2   = wave & 8191;
  int m0   = (r2 >> 4) << 4;
  int n0   = (r2 & 15) << 6;
  int lane = threadIdx.x & 31;
  int half = lane >> 4;
  int cidx = lane & 15;
  const bf16_t* wt_p = wt + (size_t)proj * DMODEL * DHEAD;

  v16bf a[4];
#pragma unroll
  for (int c = 0; c < 4; ++c) a[c] = load_a16(xb + (size_t)m0 * DHEAD + c * 32, DHEAD);

  const v8f vzero = {0.f, 0.f, 0.f, 0.f, 0.f, 0.f, 0.f, 0.f};
#pragma unroll
  for (int t = 0; t < 4; ++t) {
    v8f acc = vzero;
    int n = n0 + t * 16;
#pragma unroll
    for (int c = 0; c < 4; ++c) {
      v16bf b = load_b16(wt_p + (size_t)n * DHEAD + c * 32, DHEAD);
      acc = wmma_bf16(a[c], b, acc);
    }
    int col = n + cidx;
    int h = col >> 7, d = col & 127;
#pragma unroll
    for (int i = 0; i < 8; ++i) {
      int m = m0 + i + half * 8;
      int bidx = m >> 11;
      int s = m & 2047;
      bf16_t val = (bf16_t)acc[i];
      if (proj == 2) {
        vtb[(((size_t)(bidx * HEADS + h) * DHEAD + d) * SEQ) + s] = val;
      } else {
        bf16_t* dst = (proj == 0) ? qb : kb;
        dst[(((size_t)(bidx * HEADS + h) * SEQ + s) * DHEAD) + d] = val;
      }
    }
  }
}

// ---------------- flash attention ----------------
// grid (S/128, B*H); 8 waves/block, each wave owns 16 query rows.
// K tile (32x128) and V^T tile (128x32) staged in LDS via TDM, double-buffered.
__global__ __launch_bounds__(256) void k_attn(const bf16_t* __restrict__ qb,
                                              const bf16_t* __restrict__ kb,
                                              const bf16_t* __restrict__ vtb,
                                              bf16_t* __restrict__ attn) {
  __shared__ __align__(64) bf16_t ldsK[2 * 32 * 128];   // [buf][key][d]
  __shared__ __align__(64) bf16_t ldsV[2 * 128 * 32];   // [buf][d][key]
  __shared__ __align__(64) bf16_t lds_p[8 * 16 * 32];   // per-wave P staging

  int tid  = threadIdx.x;
  int w    = tid >> 5;
  int lane = tid & 31;
  int half = lane >> 4;
  int cidx = lane & 15;
  int bh  = blockIdx.y;
  int sq0 = blockIdx.x * 128 + w * 16;

  const bf16_t* qbase = qb + ((size_t)bh * SEQ + sq0) * DHEAD;
  const bf16_t* kbase = kb + (size_t)bh * SEQ * DHEAD;
  const bf16_t* vbase = vtb + (size_t)bh * DHEAD * SEQ;
  bf16_t* pbuf = lds_p + w * (16 * 32);

  v16bf qa[4];
#pragma unroll
  for (int c = 0; c < 4; ++c) qa[c] = load_a16(qbase + c * 32, DHEAD);

  const v8f vzero = {0.f, 0.f, 0.f, 0.f, 0.f, 0.f, 0.f, 0.f};
  v8f o[8];
  float m[8], l[8];
#pragma unroll
  for (int t = 0; t < 8; ++t) o[t] = vzero;
#pragma unroll
  for (int i = 0; i < 8; ++i) { m[i] = -1e30f; l[i] = 0.f; }

  // ---- tile staging: TDM (wave 0 issues) or cooperative copy fallback ----
#if USE_TDM
#define STAGE_TILES(N0, BUF)                                                   \
  do {                                                                         \
    if (w == 0) {                                                              \
      tdm_load_2d(lds_off(ldsK + (BUF) * 4096), kbase + (size_t)(N0) * DHEAD,  \
                  128u, 32u, 128ull);                                          \
      tdm_load_2d(lds_off(ldsV + (BUF) * 4096), vbase + (N0),                  \
                  32u, 128u, (unsigned long long)SEQ);                         \
    }                                                                          \
  } while (0)
// PEND must be a literal (SOPP immediate)
#define WAIT_TILES(PEND)                                                       \
  do { if (w == 0) __builtin_amdgcn_s_wait_tensorcnt(PEND); } while (0)
#else
#define STAGE_TILES(N0, BUF)                                                   \
  do {                                                                         \
    *(v16bf*)(ldsK + (BUF) * 4096 + tid * 16) =                                \
        *(const v16bf*)(kbase + (size_t)(N0) * DHEAD + tid * 16);              \
    int vr = tid >> 1, vh = tid & 1;                                           \
    *(v16bf*)(ldsV + (BUF) * 4096 + vr * 32 + vh * 16) =                       \
        *(const v16bf*)(vbase + (size_t)vr * SEQ + (N0) + vh * 16);            \
  } while (0)
#define WAIT_TILES(PEND) do { } while (0)
#endif

  STAGE_TILES(0, 0);                       // prologue: chunk 0 -> buffer 0

  for (int n0 = 0, it = 0; n0 < SEQ; n0 += 32, ++it) {
    int buf = it & 1;
    if (n0 + 32 < SEQ) {                   // double-buffer: prefetch next chunk
      STAGE_TILES(n0 + 32, buf ^ 1);
      WAIT_TILES(2);                       // current chunk's 2 TDM ops complete
    } else {
      WAIT_TILES(0);                       // last chunk: drain
    }
    __syncthreads();                       // data visible to all waves

    const bf16_t* kt = ldsK + buf * 4096;  // [32][128]
    const bf16_t* vt = ldsV + buf * 4096;  // [128][32]

    // scores: two 16x16 tiles covering 32 keys, K-dim = 128
    v8f sc0 = vzero, sc1 = vzero;
#pragma unroll
    for (int c = 0; c < 4; ++c) {
      v16bf b0 = load_b16(kt + c * 32, DHEAD);
      v16bf b1 = load_b16(kt + 16 * DHEAD + c * 32, DHEAD);
      sc0 = wmma_bf16(qa[c], b0, sc0);
      sc1 = wmma_bf16(qa[c], b1, sc1);
    }
    // online softmax (rows live per (half, vgpr); reduce across 16 lanes)
#pragma unroll
    for (int i = 0; i < 8; ++i) {
      float rm = fmaxf(sc0[i], sc1[i]);
#pragma unroll
      for (int mask = 1; mask < 16; mask <<= 1)
        rm = fmaxf(rm, __shfl_xor(rm, mask, 32));
      float mn = fmaxf(m[i], rm);
      float p0 = __expf(sc0[i] - mn);
      float p1 = __expf(sc1[i] - mn);
      float rs = p0 + p1;
#pragma unroll
      for (int mask = 1; mask < 16; mask <<= 1)
        rs += __shfl_xor(rs, mask, 32);
      float scale = __expf(m[i] - mn);
      l[i] = l[i] * scale + rs;
      m[i] = mn;
      int row = i + half * 8;
      pbuf[row * 32 + cidx]      = (bf16_t)p0;
      pbuf[row * 32 + 16 + cidx] = (bf16_t)p1;
#pragma unroll
      for (int t = 0; t < 8; ++t) o[t][i] *= scale;
    }
    // P (16x32) @ V (32x128); P re-read from LDS in A layout
    v16bf pa = load_a16(pbuf, 32);
#pragma unroll
    for (int t = 0; t < 8; ++t) {
      v16bf vb = load_b16(vt + (t * 16) * 32, 32);
      o[t] = wmma_bf16(pa, vb, o[t]);
    }
    __syncthreads();                       // buffer reuse safety
  }

  // normalize and store to attn[b][s][h*128+d] (bf16)
  int b = bh >> 3, h = bh & 7;
#pragma unroll
  for (int i = 0; i < 8; ++i) {
    float inv = 1.f / l[i];
    int s = sq0 + i + half * 8;
    size_t rowoff = ((size_t)(b * SEQ + s)) * DMODEL + h * DHEAD;
#pragma unroll
    for (int t = 0; t < 8; ++t)
      attn[rowoff + t * 16 + cidx] = (bf16_t)(o[t][i] * inv);
  }
}

// ---------------- output projection ----------------
__global__ __launch_bounds__(256) void k_outproj(const bf16_t* __restrict__ attn,
                                                 const bf16_t* __restrict__ wot,
                                                 const float* __restrict__ bo,
                                                 float* __restrict__ out) {
  int wave = (blockIdx.x * 256 + threadIdx.x) >> 5;
  int m0 = wave << 4;
  int lane = threadIdx.x & 31;
  int half = lane >> 4;
  int cidx = lane & 15;
  const v8f vzero = {0.f, 0.f, 0.f, 0.f, 0.f, 0.f, 0.f, 0.f};
  v8f acc[8];
#pragma unroll
  for (int t = 0; t < 8; ++t) acc[t] = vzero;
  for (int c = 0; c < DMODEL; c += 32) {
    v16bf a = load_a16(attn + (size_t)m0 * DMODEL + c, DMODEL);
#pragma unroll
    for (int t = 0; t < 8; ++t) {
      v16bf b = load_b16(wot + (size_t)(t * 16) * DMODEL + c, DMODEL);
      acc[t] = wmma_bf16(a, b, acc[t]);
    }
  }
#pragma unroll
  for (int t = 0; t < 8; ++t) {
    float bias = bo[t * 16 + cidx];
#pragma unroll
    for (int i = 0; i < 8; ++i) {
      int mrow = m0 + i + half * 8;
      out[(size_t)mrow * DHEAD + t * 16 + cidx] = acc[t][i] + bias;
    }
  }
}

extern "C" void kernel_launch(void* const* d_in, const int* in_sizes, int n_in,
                              void* d_out, int out_size, void* d_ws, size_t ws_size,
                              hipStream_t stream) {
  const float* x  = (const float*)d_in[0];
  const float* Wq = (const float*)d_in[1];
  const float* Wk = (const float*)d_in[2];
  const float* Wv = (const float*)d_in[3];
  const float* Wo = (const float*)d_in[4];
  const float* bo = (const float*)d_in[5];
  float* out = (float*)d_out;

  char* ws = (char*)d_ws;
  size_t off = 0;
  bf16_t* xb  = (bf16_t*)(ws + off); off += (size_t)ROWS * DHEAD * 2;
  bf16_t* wt  = (bf16_t*)(ws + off); off += (size_t)3 * DMODEL * DHEAD * 2;
  bf16_t* wot = (bf16_t*)(ws + off); off += (size_t)DHEAD * DMODEL * 2;
  bf16_t* qb  = (bf16_t*)(ws + off); off += (size_t)BATCH * HEADS * SEQ * DHEAD * 2;
  bf16_t* kb  = (bf16_t*)(ws + off); off += (size_t)BATCH * HEADS * SEQ * DHEAD * 2;
  bf16_t* vtb = (bf16_t*)(ws + off); off += (size_t)BATCH * HEADS * DHEAD * SEQ * 2;
  bf16_t* att = (bf16_t*)(ws + off); off += (size_t)ROWS * DMODEL * 2;
  (void)ws_size; (void)in_sizes; (void)n_in; (void)out_size;

  int nx = ROWS * DHEAD;
  k_cvt_x<<<(nx + 255) / 256, 256, 0, stream>>>(x, xb, nx);
  k_cvt_w<<<(3 * DMODEL * DHEAD) / 256, 256, 0, stream>>>(Wq, Wk, Wv, wt);
  k_cvt_wo<<<(DHEAD * DMODEL) / 256, 256, 0, stream>>>(Wo, wot);

  k_qkv<<<3072, 256, 0, stream>>>(xb, wt, qb, kb, vtb);

  k_attn<<<dim3(SEQ / 128, BATCH * HEADS), 256, 0, stream>>>(qb, kb, vtb, att);

  k_outproj<<<64, 256, 0, stream>>>(att, wot, bo, out);
}
